// MHSA_75144747810951
// MI455X (gfx1250) — matrix operand order repaired and checked by reading, not compiled, not run on hardware
//
#include <hip/hip_runtime.h>

#define D_MODEL 1024
#define HEADS   16
#define DK      64
#define SEQ     2048
#define BATCH   4
#define NROWS   (BATCH * SEQ)   // 8192

typedef __attribute__((ext_vector_type(16))) __bf16 bf16x16;
typedef __attribute__((ext_vector_type(8)))  __bf16 bf16x8;
typedef __attribute__((ext_vector_type(8)))  float  f32x8;

#define WMMA_BF16(a, b, c) \
  __builtin_amdgcn_wmma_f32_16x16x32_bf16(false, (a), false, (b), (short)0, (c), false, false)

__device__ __forceinline__ bf16x16 ld_frag2(const __bf16* p0, const __bf16* p1) {
  bf16x8 lo = *(const bf16x8*)p0;
  bf16x8 hi = *(const bf16x8*)p1;
  return __builtin_shufflevector(lo, hi, 0, 1, 2, 3, 4, 5, 6, 7,
                                 8, 9, 10, 11, 12, 13, 14, 15);
}

__device__ __forceinline__ f32x8 zero8() {
  f32x8 z;
#pragma unroll
  for (int i = 0; i < 8; ++i) z[i] = 0.0f;
  return z;
}

// Same-wave LDS ordering (no block barrier: waves have divergent loop counts).
__device__ __forceinline__ void lds_fence() {
  __asm__ volatile("s_wait_dscnt 0" ::: "memory");
}

__global__ void cvt_bf16_kernel(const float* __restrict__ in, __bf16* __restrict__ out, int n) {
  int i = blockIdx.x * blockDim.x + threadIdx.x;
  if (i < n) out[i] = (__bf16)in[i];
}

// Register-blocked WMMA GEMM: each wave computes a 32x64 tile of y = x . W^T.
// z selects Wq/Wk/Wv; RoPE applied to Q,K. N-tile (64 wide) == exactly one head.
// Q,K stored [b][h][s][dk] bf16; V stored transposed [b][h][dk][s] bf16.
__global__ void qkv_proj(const __bf16* __restrict__ xh,
                         const __bf16* __restrict__ Wq, const __bf16* __restrict__ Wk,
                         const __bf16* __restrict__ Wv,
                         __bf16* __restrict__ Qb, __bf16* __restrict__ Kb,
                         __bf16* __restrict__ Vt) {
  const int lane = threadIdx.x & 31;
  const int wave = threadIdx.x >> 5;
  const int l16  = lane & 15;
  const int half = (lane >> 4) & 1;
  const int mt = blockIdx.x;              // 32-row tile index (0..255)
  const int nt = blockIdx.y * 4 + wave;   // 64-col tile index (0..15) == head
  const int z  = blockIdx.z;
  const __bf16* W = (z == 0) ? Wq : (z == 1) ? Wk : Wv;

  const __bf16* arow0 = xh + (size_t)(mt * 32 + l16) * D_MODEL + half * 16;
  const __bf16* arow1 = arow0 + (size_t)16 * D_MODEL;
  const __bf16* brow0 = W + (size_t)(nt * 64 + l16) * D_MODEL + half * 16;

  f32x8 acc[2][4];
#pragma unroll
  for (int i = 0; i < 2; ++i)
#pragma unroll
    for (int j = 0; j < 4; ++j) acc[i][j] = zero8();

  for (int kb = 0; kb < D_MODEL; kb += 32) {
    bf16x16 a0 = ld_frag2(arow0 + kb, arow0 + kb + 8);
    bf16x16 a1 = ld_frag2(arow1 + kb, arow1 + kb + 8);
#pragma unroll
    for (int j = 0; j < 4; ++j) {
      bf16x16 bfr = *(const bf16x16*)(brow0 + (size_t)j * 16 * D_MODEL + kb);
      acc[0][j] = WMMA_BF16(a0, bfr, acc[0][j]);
      acc[1][j] = WMMA_BF16(a1, bfr, acc[1][j]);
    }
  }

  const int h = nt;                        // 64-wide N tile == one head
  if (z < 2) {
    __bf16* dst = (z == 0) ? Qb : Kb;
#pragma unroll
    for (int j = 0; j < 4; ++j) {
      const int dk = j * 16 + l16;
      const int i2 = dk & ~1;
      const float invf = __expf(-(float)i2 * (9.210340371976184f / 64.0f)); // theta^(-i2/64)
#pragma unroll
      for (int i = 0; i < 2; ++i) {
#pragma unroll
        for (int r = 0; r < 8; ++r) {
          int grow = mt * 32 + i * 16 + r + half * 8;
          int bb = grow >> 11;
          int s  = grow & (SEQ - 1);
          float v = acc[i][j][r];
          float other = __shfl_xor(v, 1, 32);   // even<->odd dk partner lane
          float ang = (float)s * invf;
          float cs = __cosf(ang), sn = __sinf(ang);
          float ov = (dk & 1) ? (other * sn + v * cs) : (v * cs - other * sn);
          dst[(((size_t)bb * HEADS + h) * SEQ + s) * DK + dk] = (__bf16)ov;
        }
      }
    }
  } else {
#pragma unroll
    for (int j = 0; j < 4; ++j) {
      const int dk = j * 16 + l16;
#pragma unroll
      for (int i = 0; i < 2; ++i) {
#pragma unroll
        for (int r = 0; r < 8; ++r) {
          int grow = mt * 32 + i * 16 + r + half * 8;
          int bb = grow >> 11;
          int s  = grow & (SEQ - 1);
          Vt[(((size_t)bb * HEADS + h) * DK + dk) * SEQ + s] = (__bf16)acc[i][j][r];
        }
      }
    }
  }
}

// Flash-attention: one wave = 16 query rows; causal online softmax; bf16 WMMA.
__global__ void attn_kernel(const __bf16* __restrict__ Qb, const __bf16* __restrict__ Kb,
                            const __bf16* __restrict__ Vt, __bf16* __restrict__ mh) {
  __shared__ alignas(64) float  sc[4][16][32];
  __shared__ alignas(64) __bf16 pb[4][16][32];
  __shared__ alignas(64) float  fact[4][16];
  __shared__ alignas(64) float  linv[4][16];

  const int lane = threadIdx.x & 31;
  const int wave = threadIdx.x >> 5;
  const int l16  = lane & 15;
  const int half = (lane >> 4) & 1;

  const int qt = blockIdx.x * 4 + wave;
  const int bh = blockIdx.y;
  const int q0 = qt * 16;

  const __bf16* Qp = Qb + (size_t)bh * SEQ * DK;
  const __bf16* Kp = Kb + (size_t)bh * SEQ * DK;
  const __bf16* Vp = Vt + (size_t)bh * DK * SEQ;

  const __bf16* qrow = Qp + (size_t)(q0 + l16) * DK + half * 16;
  bf16x16 qa0 = ld_frag2(qrow,      qrow + 8);
  bf16x16 qa1 = ld_frag2(qrow + 32, qrow + 40);

  f32x8 o0 = zero8(), o1 = zero8(), o2 = zero8(), o3 = zero8();
  float mrow = -3.0e38f, lrow = 0.0f;
  const float scale = 0.125f;   // 1/sqrt(64)
  const int qpos = q0 + l16;

  for (int kb = 0; kb <= q0; kb += 32) {
    // scores: Q(16x64) . K^T -> two 16x16 tiles (keys kb..kb+15, kb+16..kb+31)
    f32x8 s0 = zero8(), s1 = zero8();
    const __bf16* k0 = Kp + (size_t)(kb + l16) * DK + half * 16;
    const __bf16* k1 = Kp + (size_t)(kb + 16 + l16) * DK + half * 16;
    bf16x16 b00 = *(const bf16x16*)k0;
    bf16x16 b10 = *(const bf16x16*)k1;
    s0 = WMMA_BF16(qa0, b00, s0);
    s1 = WMMA_BF16(qa0, b10, s1);
    bf16x16 b01 = *(const bf16x16*)(k0 + 32);
    bf16x16 b11 = *(const bf16x16*)(k1 + 32);
    s0 = WMMA_BF16(qa1, b01, s0);
    s1 = WMMA_BF16(qa1, b11, s1);

#pragma unroll
    for (int r = 0; r < 8; ++r) {
      sc[wave][r + half * 8][l16]      = s0[r];
      sc[wave][r + half * 8][l16 + 16] = s1[r];
    }
    lds_fence();

    // online softmax for row l16 (both halves duplicate identical work)
    float mx = -3.0e38f;
#pragma unroll
    for (int c = 0; c < 32; ++c) {
      float sv = sc[wave][l16][c] * scale;
      sv = (kb + c <= qpos) ? sv : -3.0e38f;
      mx = fmaxf(mx, sv);
    }
    float mnew = fmaxf(mrow, mx);
    float fs = __expf(mrow - mnew);
    float ls = 0.0f;
#pragma unroll
    for (int c = 0; c < 32; ++c) {
      float sv = sc[wave][l16][c] * scale;
      float p  = (kb + c <= qpos) ? __expf(sv - mnew) : 0.0f;
      ls += p;
      pb[wave][l16][c] = (__bf16)p;
    }
    mrow = mnew;
    lrow = lrow * fs + ls;
    fact[wave][l16] = fs;
    lds_fence();

#pragma unroll
    for (int r = 0; r < 8; ++r) {
      float f = fact[wave][r + half * 8];
      o0[r] *= f; o1[r] *= f; o2[r] *= f; o3[r] *= f;
    }

    // P (16x32 bf16) A-fragment from LDS
    const __bf16* pr = &pb[wave][l16][half * 16];
    bf16x16 pa = ld_frag2(pr, pr + 8);

    // O += P . V  (V^T layout: column d contiguous over keys)
    {
      const __bf16* vp0 = Vp + (size_t)(0 * 16 + l16) * SEQ + kb + half * 16;
      const __bf16* vp1 = Vp + (size_t)(1 * 16 + l16) * SEQ + kb + half * 16;
      const __bf16* vp2 = Vp + (size_t)(2 * 16 + l16) * SEQ + kb + half * 16;
      const __bf16* vp3 = Vp + (size_t)(3 * 16 + l16) * SEQ + kb + half * 16;
      o0 = WMMA_BF16(pa, *(const bf16x16*)vp0, o0);
      o1 = WMMA_BF16(pa, *(const bf16x16*)vp1, o1);
      o2 = WMMA_BF16(pa, *(const bf16x16*)vp2, o2);
      o3 = WMMA_BF16(pa, *(const bf16x16*)vp3, o3);
    }
  }

  linv[wave][l16] = 1.0f / lrow;
  lds_fence();

  const int b = bh >> 4, h = bh & 15;
#pragma unroll
  for (int r = 0; r < 8; ++r) {
    int rq = r + half * 8;
    float inv = linv[wave][rq];
    size_t rowbase = ((size_t)b * SEQ + (q0 + rq)) * D_MODEL + h * DK;
    mh[rowbase +  0 + l16] = (__bf16)(o0[r] * inv);
    mh[rowbase + 16 + l16] = (__bf16)(o1[r] * inv);
    mh[rowbase + 32 + l16] = (__bf16)(o2[r] * inv);
    mh[rowbase + 48 + l16] = (__bf16)(o3[r] * inv);
  }
}

// Register-blocked WMMA GEMM: each wave computes 32x64 of out = mh . Wo^T (fp32 out).
__global__ void out_proj(const __bf16* __restrict__ mh, const __bf16* __restrict__ Wo,
                         float* __restrict__ out) {
  const int lane = threadIdx.x & 31;
  const int wave = threadIdx.x >> 5;
  const int l16  = lane & 15;
  const int half = (lane >> 4) & 1;
  const int mt = blockIdx.x;
  const int nt = blockIdx.y * 4 + wave;

  const __bf16* arow0 = mh + (size_t)(mt * 32 + l16) * D_MODEL + half * 16;
  const __bf16* arow1 = arow0 + (size_t)16 * D_MODEL;
  const __bf16* brow0 = Wo + (size_t)(nt * 64 + l16) * D_MODEL + half * 16;

  f32x8 acc[2][4];
#pragma unroll
  for (int i = 0; i < 2; ++i)
#pragma unroll
    for (int j = 0; j < 4; ++j) acc[i][j] = zero8();

  for (int kb = 0; kb < D_MODEL; kb += 32) {
    bf16x16 a0 = ld_frag2(arow0 + kb, arow0 + kb + 8);
    bf16x16 a1 = ld_frag2(arow1 + kb, arow1 + kb + 8);
#pragma unroll
    for (int j = 0; j < 4; ++j) {
      bf16x16 bfr = *(const bf16x16*)(brow0 + (size_t)j * 16 * D_MODEL + kb);
      acc[0][j] = WMMA_BF16(a0, bfr, acc[0][j]);
      acc[1][j] = WMMA_BF16(a1, bfr, acc[1][j]);
    }
  }

#pragma unroll
  for (int i = 0; i < 2; ++i) {
#pragma unroll
    for (int j = 0; j < 4; ++j) {
#pragma unroll
      for (int r = 0; r < 8; ++r) {
        out[(size_t)(mt * 32 + i * 16 + r + half * 8) * D_MODEL + nt * 64 + j * 16 + l16] =
            acc[i][j][r];
      }
    }
  }
}

extern "C" void kernel_launch(void* const* d_in, const int* in_sizes, int n_in,
                              void* d_out, int out_size, void* d_ws, size_t ws_size,
                              hipStream_t stream) {
  (void)in_sizes; (void)n_in; (void)out_size; (void)ws_size;
  const float* x  = (const float*)d_in[0];
  const float* Wq = (const float*)d_in[1];
  const float* Wk = (const float*)d_in[2];
  const float* Wv = (const float*)d_in[3];
  const float* Wo = (const float*)d_in[4];
  float* out = (float*)d_out;

  const size_t MB = 1024 * 1024;
  char* w = (char*)d_ws;
  __bf16* xh  = (__bf16*)(w);              // 16 MB  [8192][1024]
  __bf16* Wqh = (__bf16*)(w + 16 * MB);    //  2 MB
  __bf16* Wkh = (__bf16*)(w + 18 * MB);    //  2 MB
  __bf16* Wvh = (__bf16*)(w + 20 * MB);    //  2 MB
  __bf16* Woh = (__bf16*)(w + 22 * MB);    //  2 MB
  __bf16* Qb  = (__bf16*)(w + 24 * MB);    // 16 MB  [b][h][s][dk]
  __bf16* Kb  = (__bf16*)(w + 40 * MB);    // 16 MB  [b][h][s][dk]
  __bf16* Vt  = (__bf16*)(w + 56 * MB);    // 16 MB  [b][h][dk][s]
  __bf16* mhb = (__bf16*)(w + 72 * MB);    // 16 MB  [8192][1024]

  const int nx = NROWS * D_MODEL;
  const int nw = D_MODEL * D_MODEL;
  cvt_bf16_kernel<<<(nx + 255) / 256, 256, 0, stream>>>(x,  xh,  nx);
  cvt_bf16_kernel<<<(nw + 255) / 256, 256, 0, stream>>>(Wq, Wqh, nw);
  cvt_bf16_kernel<<<(nw + 255) / 256, 256, 0, stream>>>(Wk, Wkh, nw);
  cvt_bf16_kernel<<<(nw + 255) / 256, 256, 0, stream>>>(Wv, Wvh, nw);
  cvt_bf16_kernel<<<(nw + 255) / 256, 256, 0, stream>>>(Wo, Woh, nw);

  qkv_proj<<<dim3(NROWS / 32, D_MODEL / 64 / 4, 3), 128, 0, stream>>>(
      xh, Wqh, Wkh, Wvh, Qb, Kb, Vt);

  attn_kernel<<<dim3(SEQ / 64, BATCH * HEADS), 128, 0, stream>>>(Qb, Kb, Vt, mhb);

  out_proj<<<dim3(NROWS / 32, D_MODEL / 64 / 4), 128, 0, stream>>>(mhb, Woh, out);
}
